// MultiHeadAttention_20770461843824
// MI455X (gfx1250) — matrix operand order repaired and checked by reading, not compiled, not run on hardware
//
#include <hip/hip_runtime.h>

typedef __attribute__((ext_vector_type(16))) _Float16 v16h;
typedef __attribute__((ext_vector_type(8)))  _Float16 v8h;
typedef __attribute__((ext_vector_type(8)))  float    v8f;

union V16 { v16h v; v8h h[2]; };

constexpr int B  = 4;
constexpr int S  = 2048;
constexpr int D  = 1024;
constexpr int H  = 16;
constexpr int DH = 64;
constexpr int KC = 64;          // keys per chunk

// 1/sqrt(DH) * log2(e): scores end up in log2 domain -> exp2 softmax
#define QSCALE 0.18033688011112042f

#if __has_builtin(__builtin_amdgcn_tensor_load_to_lds)
#define HAVE_TDM 1
#else
#define HAVE_TDM 0
#endif

__device__ __forceinline__ float fast_exp2(float x) {
#if __has_builtin(__builtin_amdgcn_exp2f)
    return __builtin_amdgcn_exp2f(x);
#else
    return exp2f(x);
#endif
}

// max-reduce across the 16-lane row this lane belongs to (VALU crossbar, no LDS)
__device__ __forceinline__ float rowmax16(float v) {
#if __has_builtin(__builtin_amdgcn_permlane16)
    int iv;
    iv = __builtin_bit_cast(int, v);
    v  = fmaxf(v, __builtin_bit_cast(float, __builtin_amdgcn_permlane16(iv, iv, 0x67452301u, 0xefcdab89u, false, false)));
    iv = __builtin_bit_cast(int, v);
    v  = fmaxf(v, __builtin_bit_cast(float, __builtin_amdgcn_permlane16(iv, iv, 0x54761032u, 0xdcfe98bau, false, false)));
    iv = __builtin_bit_cast(int, v);
    v  = fmaxf(v, __builtin_bit_cast(float, __builtin_amdgcn_permlane16(iv, iv, 0x32107654u, 0xba98fedcu, false, false)));
    iv = __builtin_bit_cast(int, v);
    v  = fmaxf(v, __builtin_bit_cast(float, __builtin_amdgcn_permlane16(iv, iv, 0xfedcba98u, 0x76543210u, false, false)));
    return v;
#else
    for (int xm = 1; xm < 16; xm <<= 1) v = fmaxf(v, __shfl_xor(v, xm, 32));
    return v;
#endif
}

// CDNA5 async copy: 16B per lane, memory -> LDS, no VGPR data, ASYNCcnt-tracked
__device__ __forceinline__ void async_b128(const _Float16* g, _Float16* l) {
    const unsigned lds = (unsigned)(size_t)l;    // low 32 bits of flat = LDS offset
    asm volatile("global_load_async_to_lds_b128 %0, %1, off"
                 :: "v"(lds), "v"(g) : "memory");
}

__device__ __forceinline__ void wait_async0() {
#if __has_builtin(__builtin_amdgcn_s_wait_asynccnt)
    __builtin_amdgcn_s_wait_asynccnt(0);
#else
    asm volatile("s_wait_asynccnt 0x0" ::: "memory");
#endif
}

__device__ __forceinline__ void wait_tensor0() {
#if __has_builtin(__builtin_amdgcn_s_wait_tensorcnt)
    __builtin_amdgcn_s_wait_tensorcnt(0);
#else
    asm volatile("s_wait_tensorcnt 0x0" ::: "memory");
#endif
}

#if HAVE_TDM
// TDM: DMA a 64x64 f16 row-major tile (row stride = `stride0` elements) into
// LDS, hardware-padding each 128B row by 16B to keep the bank-conflict-free
// layout. D# per CDNA5 ISA Ch8: group0 {count, lds_addr, gaddr}, group1
// {data_size/pad, tensor dims, tile dims, strides}; groups 2/3 zero (2-D).
__device__ __forceinline__ void tdm_load_tile64(const _Float16* gaddr,
                                                unsigned lds_addr,
                                                unsigned stride0 /*elements*/) {
    typedef __attribute__((ext_vector_type(4))) unsigned u32x4;
    typedef __attribute__((ext_vector_type(8))) int      i32x8;
    typedef __attribute__((ext_vector_type(4))) int      i32x4;
    const unsigned long long ga = (unsigned long long)(size_t)gaddr;
    u32x4 g0 = { 1u,                                   // count=1, user desc
                 lds_addr,                             // LDS dest (bytes)
                 (unsigned)ga,                         // global addr [31:0]
                 ((unsigned)(ga >> 32) & 0x01FFFFFFu)  // global addr [56:32]
                   | 0x80000000u };                    // type=2 (image)
    // w0: data_size=2B (1<<16) | pad_enable (1<<20) | pad_interval=32dw (4<<22)
    //     | pad_amount=4dw (3<<25)  => pad 16B after each 128B row
    i32x8 g1 = { (int)0x07110000,
                 (int)(64u << 16),        // tensor_dim0 = 64
                 (int)(64u << 16),        // tensor_dim1 = 64
                 (int)(64u << 16),        // tile_dim0  = 64
                 64,                      // tile_dim1  = 64, tile_dim2 = 0
                 (int)stride0,            // tensor_dim0_stride [31:0]
                 0, 0 };
    i32x4 z4 = { 0, 0, 0, 0 };
    i32x8 z8 = { 0, 0, 0, 0, 0, 0, 0, 0 };
    __builtin_amdgcn_tensor_load_to_lds(g0, g1, z4, z4, z8, 0);
}
#endif

// ---------------------------------------------------------------------------
// Kernel 1: per-head QKV projection, fp32 -> fp16.
//   Q [B,H,S,DH] with QSCALE folded in; K [B,H,S,DH]; V transposed [B,H,DH,S].
// ---------------------------------------------------------------------------
__global__ __launch_bounds__(256) void qkv_proj(
    const float* __restrict__ x,
    const float* __restrict__ Wq, const float* __restrict__ bq,
    const float* __restrict__ Wk, const float* __restrict__ bk,
    const float* __restrict__ Wv, const float* __restrict__ bv,
    _Float16* __restrict__ Qh, _Float16* __restrict__ Kh, _Float16* __restrict__ Vt)
{
    const int i   = blockIdx.x * 256 + threadIdx.x;   // [0, B*S*H*DH)
    const int e   = i & (DH - 1);
    const int h   = (i >> 6) & (H - 1);
    const int tok = i >> 10;                          // b*S + s
    const int b   = tok / S;
    const int s   = tok - b * S;

    const float* xr = x  + (size_t)tok * D + h * DH;
    const float* wq = Wq + ((size_t)h * DH + e) * DH;
    const float* wk = Wk + ((size_t)h * DH + e) * DH;
    const float* wv = Wv + ((size_t)h * DH + e) * DH;

    float aq = 0.f, ak = 0.f, av = 0.f;
    #pragma unroll 8
    for (int d = 0; d < DH; ++d) {
        const float xv = xr[d];
        aq = fmaf(xv, wq[d], aq);
        ak = fmaf(xv, wk[d], ak);
        av = fmaf(xv, wv[d], av);
    }
    aq += bq[h * DH + e];
    ak += bk[h * DH + e];
    av += bv[h * DH + e];

    const size_t rowmaj = ((size_t)(b * H + h) * S + s) * DH + e;
    Qh[rowmaj] = (_Float16)(aq * QSCALE);
    Kh[rowmaj] = (_Float16)ak;
    Vt[((size_t)(b * H + h) * DH + e) * S + s] = (_Float16)av;
}

// ---------------------------------------------------------------------------
// Kernel 2: flash attention. 256 thr = 8 waves; each wave owns 16 query rows
// (128/block) of one (b,h). 64-key chunks double-buffered in LDS:
//   K tile via TDM (tensor_load_to_lds, wave 0 issues, TENSORcnt),
//   V tile via per-lane global_load_async_to_lds_b128 (ASYNCcnt).
// Per chunk per wave: 8 wmma QK^T, 2 wmma row-sum (P*ones), 8 wmma PV.
// ---------------------------------------------------------------------------
__global__ __launch_bounds__(256) void flash_attn(
    const _Float16* __restrict__ Qh, const _Float16* __restrict__ Kh,
    const _Float16* __restrict__ Vt, float* __restrict__ out)
{
    __shared__ _Float16 Ks[2][KC][72];     // 64 keys  x 64 feats (+16B pad), x2 buf
    __shared__ _Float16 Vs[2][DH][72];     // 64 feats x 64 keys  (+16B pad), x2 buf
    __shared__ _Float16 Ps[8][16][72];     // per-wave P tile 16q x 64k (+pad)

    const int t    = threadIdx.x;
    const int wave = t >> 5;
    const int lane = t & 31;
    const int hl   = lane >> 4;            // 16-lane half
    const int ln   = lane & 15;

    const int b  = blockIdx.z;
    const int h  = blockIdx.y;
    const int q0 = blockIdx.x * 128 + wave * 16;

    const size_t bh = (size_t)(b * H + h);
    const _Float16* Qbase = Qh + bh * S * DH;
    const _Float16* Kbase = Kh + bh * S * DH;
    const _Float16* Vbase = Vt + bh * DH * S;

    // ---- K tile: one TDM descriptor (wave 0); fallback: per-lane async
    auto issueK = [&](int buf, int j0) {
#if HAVE_TDM
        if (wave == 0)
            tdm_load_tile64(Kbase + (size_t)j0 * DH,
                            (unsigned)(size_t)&Ks[buf][0][0], DH);
#else
        #pragma unroll
        for (int p = 0; p < 2; ++p) {
            const int idx = t + p * 256;
            const int r = idx >> 3, cl = (idx & 7) * 8;
            async_b128(Kbase + (size_t)(j0 + r) * DH + cl, &Ks[buf][r][cl]);
        }
#endif
    };
    // ---- V tile: per-lane async (strided rows of Vt)
    auto issueV = [&](int buf, int j0) {
        #pragma unroll
        for (int p = 0; p < 2; ++p) {
            const int idx = t + p * 256;
            const int r = idx >> 3, cl = (idx & 7) * 8;
            async_b128(Vbase + (size_t)r * S + j0 + cl, &Vs[buf][r][cl]);
        }
    };
    auto wait_tiles = [&]() {
        wait_async0();
#if HAVE_TDM
        if (wave == 0) wait_tensor0();
#endif
    };

    // ---- Q fragments (A-layout 16x32): row = ln, K split by lane half
    v16h qa[2];
    {
        const _Float16* qrow = Qbase + (size_t)(q0 + ln) * DH;
        const int kb = hl * 8;
        #pragma unroll
        for (int f = 0; f < 2; ++f) {
            const _Float16* p = qrow + f * 32 + kb;
            V16 tmp;
            tmp.h[0] = *(const v8h*)p;
            tmp.h[1] = *(const v8h*)(p + 16);
            qa[f] = tmp.v;
        }
    }

    // ---- all-ones B fragment: rowsum(P) = P x ones via WMMA
    v16h ones;
    #pragma unroll
    for (int i = 0; i < 16; ++i) ones[i] = (_Float16)1.0f;

    v8f   o[4] = {};
    float m[8], l[8];
    #pragma unroll
    for (int r = 0; r < 8; ++r) { m[r] = -1e30f; l[r] = 0.f; }

    issueK(0, 0);
    issueV(0, 0);
    wait_tiles();
    __syncthreads();

    for (int chunk = 0; chunk < S / KC; ++chunk) {
        const int buf = chunk & 1;
        if (chunk + 1 < S / KC) {          // overlap next chunk's DMA w/ compute
            issueK(buf ^ 1, (chunk + 1) * KC);
            issueV(buf ^ 1, (chunk + 1) * KC);
        }

        // ---- S = Q K^T : four 16x16 col-tiles, K-dim split over DH=64
        v8f c[4] = {};
        #pragma unroll
        for (int jh = 0; jh < 4; ++jh) {
            #pragma unroll
            for (int f = 0; f < 2; ++f) {
                const _Float16* p = &Ks[buf][jh * 16 + ln][f * 32 + hl * 8];
                V16 bm;
                bm.h[0] = *(const v8h*)p;
                bm.h[1] = *(const v8h*)(p + 16);
                c[jh] = __builtin_amdgcn_wmma_f32_16x16x32_f16(
                    false, qa[f], false, bm.v, (short)0, c[jh], false, false);
            }
        }

        // ---- online softmax (log2 domain): row max via permlane butterfly
        float mc[8], alpha[8];
        #pragma unroll
        for (int r = 0; r < 8; ++r)
            mc[r] = fmaxf(fmaxf(c[0][r], c[1][r]), fmaxf(c[2][r], c[3][r]));
        #pragma unroll
        for (int r = 0; r < 8; ++r) mc[r] = rowmax16(mc[r]);
        #pragma unroll
        for (int r = 0; r < 8; ++r) {
            const float mn = fmaxf(m[r], mc[r]);
            alpha[r] = fast_exp2(m[r] - mn);
            m[r]     = mn;
            #pragma unroll
            for (int jh = 0; jh < 4; ++jh)
                c[jh][r] = fast_exp2(c[jh][r] - mn);
        }
        #pragma unroll
        for (int eo = 0; eo < 4; ++eo)
            #pragma unroll
            for (int r = 0; r < 8; ++r) o[eo][r] *= alpha[r];

        // ---- P: C-layout regs -> LDS -> reload as two A-fragments
        #pragma unroll
        for (int jh = 0; jh < 4; ++jh)
            #pragma unroll
            for (int r = 0; r < 8; ++r)
                Ps[wave][hl * 8 + r][jh * 16 + ln] = (_Float16)c[jh][r];
        asm volatile("" ::: "memory");
        v16h pa[2];
        {
            const int kb = hl * 8;
            #pragma unroll
            for (int f = 0; f < 2; ++f) {
                const _Float16* pp = &Ps[wave][ln][f * 32 + kb];
                V16 tmp;
                tmp.h[0] = *(const v8h*)pp;
                tmp.h[1] = *(const v8h*)(pp + 16);
                pa[f] = tmp.v;
            }
        }

        // ---- l update via WMMA row-sum: cs = P x ones
        v8f cs = {};
        cs = __builtin_amdgcn_wmma_f32_16x16x32_f16(false, pa[0], false, ones, (short)0, cs, false, false);
        cs = __builtin_amdgcn_wmma_f32_16x16x32_f16(false, pa[1], false, ones, (short)0, cs, false, false);
        #pragma unroll
        for (int r = 0; r < 8; ++r) l[r] = l[r] * alpha[r] + cs[r];

        // ---- O += P V : 4 col-tiles x 2 K-splits
        #pragma unroll
        for (int f = 0; f < 2; ++f) {
            #pragma unroll
            for (int eo = 0; eo < 4; ++eo) {
                const _Float16* vp = &Vs[buf][eo * 16 + ln][f * 32 + hl * 8];
                V16 bm;
                bm.h[0] = *(const v8h*)vp;
                bm.h[1] = *(const v8h*)(vp + 16);
                o[eo] = __builtin_amdgcn_wmma_f32_16x16x32_f16(
                    false, pa[f], false, bm.v, (short)0, o[eo], false, false);
            }
        }

        wait_tiles();         // next chunk's tiles landed (hidden by compute)
        __syncthreads();      // one barrier per iteration
    }

    // ---- epilogue: normalize, write fp32 output [B,S,D]
    #pragma unroll
    for (int r = 0; r < 8; ++r) {
        const float inv = 1.f / l[r];
        const int   row = hl * 8 + r;
        float* op = out + ((size_t)b * S + (q0 + row)) * D + h * DH;
        #pragma unroll
        for (int eo = 0; eo < 4; ++eo)
            op[eo * 16 + ln] = o[eo][r] * inv;
    }
}

// ---------------------------------------------------------------------------
extern "C" void kernel_launch(void* const* d_in, const int* in_sizes, int n_in,
                              void* d_out, int out_size, void* d_ws, size_t ws_size,
                              hipStream_t stream) {
    const float* x  = (const float*)d_in[0];
    const float* Wq = (const float*)d_in[1];
    const float* bq = (const float*)d_in[2];
    const float* Wk = (const float*)d_in[3];
    const float* bk = (const float*)d_in[4];
    const float* Wv = (const float*)d_in[5];
    const float* bv = (const float*)d_in[6];
    float* out = (float*)d_out;

    const size_t per = (size_t)B * H * S * DH;   // elements per tensor
    _Float16* Qh = (_Float16*)d_ws;
    _Float16* Kh = Qh + per;
    _Float16* Vt = Kh + per;

    qkv_proj<<<(B * S * H * DH) / 256, 256, 0, stream>>>(
        x, Wq, bq, Wk, bk, Wv, bv, Qh, Kh, Vt);
    flash_attn<<<dim3(S / 128, H, B), 256, 0, stream>>>(Qh, Kh, Vt, out);
}
